// CNN_GAT_60327110640272
// MI455X (gfx1250) — matrix operand order repaired and checked by reading, not compiled, not run on hardware
//
#include <hip/hip_runtime.h>
#include <math.h>

// ---------------- problem constants ----------------
#define BB   16
#define EE   128
#define WW   256
#define HID  64
#define HOR  12
#define NN   (BB*EE)        // 2048
#define WIN  (WW+HOR)       // 268 circular window

typedef __attribute__((ext_vector_type(16))) _Float16 v16h;
typedef __attribute__((ext_vector_type(8)))  float    v8f;

// ---------------- WMMA fragment helpers (wave32, 16x16x32 f16) ----------------
// A: 16x32 tile, row-major source, stride lda
static __device__ inline v16h load_a_frag(const _Float16* A, int row0, int k0, int lda) {
  int lane = threadIdx.x & 31;
  int m  = lane & 15;
  int hs = (lane >> 4) * 16;
  const _Float16* p = A + (size_t)(row0 + m) * lda + k0;
  v16h a;
#pragma unroll
  for (int i = 0; i < 16; ++i) a[i] = p[hs + i];
  return a;
}
// B stored transposed (Bt is N x K row-major, i.e. weight (out,in)): Bmat[k][n] = Bt[n][k]
static __device__ inline v16h load_b_t(const _Float16* Bt, int n0, int k0, int ldt) {
  int lane = threadIdx.x & 31;
  int n  = lane & 15;
  int kb = (lane >> 4) * 16;
  const _Float16* p = Bt + (size_t)(n0 + n) * ldt + k0 + kb;
  v16h b;
#pragma unroll
  for (int i = 0; i < 16; ++i) b[i] = p[i];
  return b;
}
// B stored natural (K x N row-major, stride ldb)
static __device__ inline v16h load_b_n(const _Float16* Bm, int k0, int n0, int ldb) {
  int lane = threadIdx.x & 31;
  int n  = lane & 15;
  int kb = (lane >> 4) * 16;
  const _Float16* p = Bm + (size_t)(k0 + kb) * ldb + n0 + n;
  v16h b;
#pragma unroll
  for (int i = 0; i < 16; ++i) b[i] = p[(size_t)i * ldb];
  return b;
}

#define WMMA_F16(acc, a, b) \
  __builtin_amdgcn_wmma_f32_16x16x32_f16(false, (a), false, (b), (short)0, (acc), false, false)

// ---------------- prep: per-batch mean / std ----------------
__global__ void stats_kernel(const float* __restrict__ x, float* meanb, float* stdb) {
  int b = blockIdx.x, t = threadIdx.x;
  float v = x[b * WW + t];
  __shared__ float s1[WW], s2[WW];
  s1[t] = v; s2[t] = v * v; __syncthreads();
  for (int s = WW / 2; s > 0; s >>= 1) {
    if (t < s) { s1[t] += s1[t + s]; s2[t] += s2[t + s]; }
    __syncthreads();
  }
  if (t == 0) {
    float m = s1[0] / (float)WW;
    meanb[b] = m;
    stdb[b]  = sqrtf(s2[0] / (float)WW - m * m);
  }
}

// ---------------- prep: conv1(relu,bn) -> conv2(1x1) -> affine -> xf / xf16 / window ----------------
__global__ void prep_xf_kernel(const float* __restrict__ x, const float* meanb, const float* stdb,
                               const float* pw1, const float* pb1,
                               const float* bng, const float* bnb, const float* bnm, const float* bnv,
                               const float* pw2, const float* pb2,
                               const float* aw, const float* ab,
                               float* xf, _Float16* xf16, float* xc) {
  int b = blockIdx.x, tid = threadIdx.x;
  __shared__ float xn[WW];
  __shared__ float h0[32 * WW];
  float m = meanb[b], sd = stdb[b];
  xn[tid] = (x[b * WW + tid] - m) / sd;
  __syncthreads();
  for (int idx = tid; idx < 32 * WW; idx += blockDim.x) {
    int c = idx >> 8, t = idx & 255;
    float acc = pb1[c];
#pragma unroll
    for (int k = 0; k < 5; ++k) {
      int tt = t + k - 2;
      if (tt >= 0 && tt < WW) acc += pw1[c * 5 + k] * xn[tt];
    }
    acc = fmaxf(acc, 0.f);
    acc = (acc - bnm[c]) / sqrtf(bnv[c] + 1e-5f) * bng[c] + bnb[c];
    h0[idx] = acc;
  }
  __syncthreads();
  for (int idx = tid; idx < EE * WW; idx += blockDim.x) {
    int e = idx >> 8, t = idx & 255;
    float acc = pb2[e];
    for (int c = 0; c < 32; ++c) acc += pw2[e * 32 + c] * h0[c * WW + t];
    acc = acc * aw[e] + ab[e];
    int row = b * EE + e;
    xf[(size_t)row * WW + t]   = acc;
    xf16[(size_t)row * WW + t] = (_Float16)acc;
    xc[(size_t)row * WIN + t]  = acc;
  }
}

// ---------------- generic f32 -> f16 cast ----------------
__global__ void cast16_kernel(const float* __restrict__ s, _Float16* d, int n) {
  int i = blockIdx.x * blockDim.x + threadIdx.x;
  if (i < n) d[i] = (_Float16)s[i];
}

// ---------------- q/k projection: (2048x256)x(256x32), WMMA ----------------
__global__ void gemm_qk_kernel(const _Float16* __restrict__ xf16,
                               const _Float16* __restrict__ qw16, const _Float16* __restrict__ kw16,
                               const float* qb, const float* kb,
                               _Float16* q16, _Float16* k16) {
  int tm = blockIdx.x, tn = blockIdx.y;
  const _Float16* Bt = blockIdx.z ? kw16 : qw16;
  const float* bias  = blockIdx.z ? kb : qb;
  _Float16* outp     = blockIdx.z ? k16 : q16;
  v8f acc = {};
  for (int k0 = 0; k0 < WW; k0 += 32) {
    v16h a = load_a_frag(xf16, tm * 16, k0, WW);
    v16h bfr = load_b_t(Bt, tn * 16, k0, WW);
    acc = WMMA_F16(acc, a, bfr);
  }
  int lane = threadIdx.x & 31;
  int n = lane & 15, mo = (lane >> 4) * 8;
  float bv = bias[tn * 16 + n];
#pragma unroll
  for (int r = 0; r < 8; ++r)
    outp[(size_t)(tm * 16 + mo + r) * 32 + tn * 16 + n] = (_Float16)(acc[r] + bv);
}

// ---------------- adjacency logits: per-batch q(128x32) x k^T, WMMA ----------------
__global__ void gemm_adj_kernel(const _Float16* __restrict__ q16, const _Float16* __restrict__ k16,
                                float* logits) {
  int tm = blockIdx.x, tn = blockIdx.y, b = blockIdx.z;
  const _Float16* A  = q16 + (size_t)b * EE * 32;
  const _Float16* Kt = k16 + (size_t)b * EE * 32;
  v16h a  = load_a_frag(A, tm * 16, 0, 32);
  v16h bf = load_b_t(Kt, tn * 16, 0, 32);
  v8f acc = {};
  acc = WMMA_F16(acc, a, bf);
  int lane = threadIdx.x & 31;
  int n = lane & 15, mo = (lane >> 4) * 8;
  float* C = logits + (size_t)b * EE * EE;
#pragma unroll
  for (int r = 0; r < 8; ++r)
    C[(size_t)(tm * 16 + mo + r) * EE + tn * 16 + n] = acc[r];
}

// ---------------- row softmax over j; emit adj_soft (output) + adj_bin ----------------
__global__ void adj_softmax_kernel(const float* __restrict__ logits, float* adj_soft, float* adjbin) {
  int row = blockIdx.x;      // b*128+i
  int j = threadIdx.x;       // 128
  float v = logits[(size_t)row * EE + j];
  __shared__ float red[EE];
  red[j] = v; __syncthreads();
  for (int s = EE / 2; s > 0; s >>= 1) { if (j < s) red[j] = fmaxf(red[j], red[j + s]); __syncthreads(); }
  float mx = red[0]; __syncthreads();
  float ex = expf(v - mx);
  red[j] = ex; __syncthreads();
  for (int s = EE / 2; s > 0; s >>= 1) { if (j < s) red[j] += red[j + s]; __syncthreads(); }
  float p = ex / red[0];
  adj_soft[(size_t)row * EE + j] = p;
  adjbin[(size_t)row * EE + j]   = (p > 0.5f) ? 1.0f : 0.0f;
}

// ---------------- Q0 init ----------------
__global__ void init_q_kernel(float* Q) {
  int n = blockIdx.x * blockDim.x + threadIdx.x;
  if (n < NN) { Q[n * 3] = 1.f; Q[n * 3 + 1] = 0.f; Q[n * 3 + 2] = 0.f; }
}

// ---------------- per-(b,e) 3-layer CNN, fully fused in LDS ----------------
__global__ void cnn_kernel(const float* __restrict__ xc,
                           const float* __restrict__ w1, const float* __restrict__ b1,
                           const float* __restrict__ w2, const float* __restrict__ b2,
                           const float* __restrict__ w3, const float* __restrict__ b3,
                           float* h, _Float16* h16, int t) {
  int n = blockIdx.x;           // b*E + e
  int e = n & (EE - 1);
  int tid = threadIdx.x;        // 256
  __shared__ float xw[WW];
  __shared__ float h1[64 * 128];
  __shared__ float h2[64 * 64];
  xw[tid] = xc[(size_t)n * WIN + t + tid];
  __syncthreads();
  const float* W1 = w1 + e * 128;
  const float* B1 = b1 + e * 64;
  for (int idx = tid; idx < 64 * 128; idx += 256) {
    int c = idx >> 7, tt = idx & 127;
    h1[idx] = tanhf(W1[c * 2] * xw[2 * tt] + W1[c * 2 + 1] * xw[2 * tt + 1] + B1[c]);
  }
  __syncthreads();
  const float* W2 = w2 + e * 512;
  const float* B2 = b2 + e * 64;
  for (int idx = tid; idx < 64 * 64; idx += 256) {
    int oc = idx >> 6, tt = idx & 63;
    int g = oc >> 2;
    float acc = B2[oc];
#pragma unroll
    for (int i2 = 0; i2 < 4; ++i2) {
      acc += W2[oc * 8 + i2 * 2]     * h1[(g * 4 + i2) * 128 + 2 * tt];
      acc += W2[oc * 8 + i2 * 2 + 1] * h1[(g * 4 + i2) * 128 + 2 * tt + 1];
    }
    h2[idx] = tanhf(acc);
  }
  __syncthreads();
  if (tid < 64) {
    const float* W3 = w3 + e * 64;
    float acc = b3[e];
    for (int c = 0; c < 64; ++c) acc += W3[c] * h2[c * 64 + tid];
    float v = tanhf(acc);
    h[(size_t)n * HID + tid]   = v;
    h16[(size_t)n * HID + tid] = (_Float16)v;
  }
}

// ---------------- GAT gl/gr: (2048x64)x(64x64), WMMA ----------------
__global__ void gemm_gat_kernel(const _Float16* __restrict__ h16,
                                const _Float16* __restrict__ wl16, const _Float16* __restrict__ wr16,
                                const float* bl, const float* br,
                                float* gl, float* gr, _Float16* gr16) {
  int tm = blockIdx.x, tn = blockIdx.y, z = blockIdx.z;
  const _Float16* Bt = z ? wr16 : wl16;
  const float* bias  = z ? br : bl;
  float* outp        = z ? gr : gl;
  v8f acc = {};
  for (int k0 = 0; k0 < HID; k0 += 32) {
    v16h a  = load_a_frag(h16, tm * 16, k0, HID);
    v16h bf = load_b_t(Bt, tn * 16, k0, HID);
    acc = WMMA_F16(acc, a, bf);
  }
  int lane = threadIdx.x & 31;
  int n = lane & 15, mo = (lane >> 4) * 8;
  float bv = bias[tn * 16 + n];
#pragma unroll
  for (int r = 0; r < 8; ++r) {
    int row = tm * 16 + mo + r, col = tn * 16 + n;
    float v = acc[r] + bv;
    outp[(size_t)row * HID + col] = v;
    if (z) gr16[(size_t)row * HID + col] = (_Float16)v;
  }
}

// ---------------- GAT edge scores + masked softmax -> a16 ----------------
__global__ void gat_e_kernel(const float* __restrict__ gl, const float* __restrict__ gr,
                             const float* __restrict__ attn, const float* __restrict__ adjbin,
                             _Float16* a16) {
  int row = blockIdx.x;       // b*128 + i
  int b = row >> 7;
  int j = threadIdx.x;        // 128
  __shared__ float gri[HID];
  __shared__ float at[HID];
  __shared__ float red[EE];
  if (j < HID) { gri[j] = gr[(size_t)row * HID + j]; at[j] = attn[j]; }
  __syncthreads();
  const float* glj = gl + (size_t)(b * EE + j) * HID;
  float s = 0.f;
  for (int f = 0; f < HID; ++f) {
    float v = gri[f] + glj[f];
    v = (v > 0.f) ? v : 0.2f * v;   // leaky_relu(0.2)
    s += v * at[f];
  }
  if (adjbin[(size_t)row * EE + j] == 0.f) s = -1e20f;
  red[j] = s; __syncthreads();
  for (int st = EE / 2; st > 0; st >>= 1) { if (j < st) red[j] = fmaxf(red[j], red[j + st]); __syncthreads(); }
  float mx = red[0]; __syncthreads();
  float ex = expf(s - mx);
  red[j] = ex; __syncthreads();
  for (int st = EE / 2; st > 0; st >>= 1) { if (j < st) red[j] += red[j + st]; __syncthreads(); }
  a16[(size_t)row * EE + j] = (_Float16)(ex / red[0]);
}

// ---------------- h_temp = a @ gr per batch: (128x128)x(128x64), WMMA ----------------
__global__ void gemm_ht_kernel(const _Float16* __restrict__ a16, const _Float16* __restrict__ gr16,
                               float* ht) {
  int tm = blockIdx.x, tn = blockIdx.y, b = blockIdx.z;
  const _Float16* A  = a16  + (size_t)b * EE * EE;
  const _Float16* Bg = gr16 + (size_t)b * EE * HID;
  v8f acc = {};
  for (int k0 = 0; k0 < EE; k0 += 32) {
    v16h a  = load_a_frag(A, tm * 16, k0, EE);
    v16h bf = load_b_n(Bg, k0, tn * 16, HID);
    acc = WMMA_F16(acc, a, bf);
  }
  int lane = threadIdx.x & 31;
  int n = lane & 15, mo = (lane >> 4) * 8;
  float* C = ht + (size_t)b * EE * HID;
#pragma unroll
  for (int r = 0; r < 8; ++r)
    C[(size_t)(tm * 16 + mo + r) * HID + tn * 16 + n] = acc[r];
}

// ---------------- combine: h + tanh(h_temp) + tanh(Q ce) ----------------
__global__ void combine_kernel(const float* __restrict__ h, const float* __restrict__ ht,
                               const float* __restrict__ Qc,
                               const float* __restrict__ cew, const float* __restrict__ ceb,
                               float* hn) {
  int idx = blockIdx.x * blockDim.x + threadIdx.x;   // < 2048*64
  int n = idx >> 6, f = idx & 63;
  float hc = Qc[n * 3] * cew[f * 3] + Qc[n * 3 + 1] * cew[f * 3 + 1] + Qc[n * 3 + 2] * cew[f * 3 + 2] + ceb[f];
  hn[idx] = h[idx] + tanhf(ht[idx]) + tanhf(hc);
}

// ---------------- x_pred + P (5 simultaneous dots, LDS tree-reduce) ----------------
__global__ void pred_p_kernel(const float* __restrict__ hn,
                              const float* __restrict__ outw, const float* __restrict__ outb,
                              const float* __restrict__ cdw, const float* __restrict__ cdb,
                              float* xpa, float* xc, float* P, int t) {
  int n = blockIdx.x;           // 2048
  int f = threadIdx.x;          // 64
  int e = n & (EE - 1);
  float hv = hn[(size_t)n * HID + f];
  __shared__ float red[5][HID];
  red[0][f] = hv * outw[e * HID + f];
  red[1][f] = hv * cdw[0 * HID + f];
  red[2][f] = hv * cdw[1 * HID + f];
  red[3][f] = hv * cdw[2 * HID + f];
  red[4][f] = hv * cdw[3 * HID + f];
  __syncthreads();
  for (int s = HID / 2; s > 0; s >>= 1) {
    if (f < s) {
#pragma unroll
      for (int q = 0; q < 5; ++q) red[q][f] += red[q][f + s];
    }
    __syncthreads();
  }
  if (f == 0) {
    float xp = red[0][0] + outb[e];
    xpa[(size_t)t * NN + n]        = xp;
    xc[(size_t)n * WIN + WW + t]   = xp;   // append to circular window
  }
  if (f < 4) {
    float pv = red[1 + f][0] + cdb[f];
    P[n * 4 + f] = 1.f / (1.f + expf(-pv));
  }
}

// ---------------- mechanism recurrence (ping-pong Q) ----------------
__global__ void mech_kernel(const float* __restrict__ Qo, float* Qn,
                            const float* __restrict__ P, const float* __restrict__ adjbin,
                            const float* __restrict__ noise_t) {
  int n = blockIdx.x * blockDim.x + threadIdx.x;
  if (n >= NN) return;
  int b = n >> 7, i = n & (EE - 1);
  float H = Qo[n * 3], Ee = Qo[n * 3 + 1], V = Qo[n * 3 + 2];
  const float* arow = adjbin + (size_t)(b * EE + i) * EE;
  const float* Eb = Qo + (size_t)b * EE * 3;
  float ec = 0.f;
  for (int j = 0; j < EE; ++j) ec += arow[j] * Eb[j * 3 + 1];
  float al = P[n * 4], be = P[n * 4 + 1], ga = P[n * 4 + 2], et = P[n * 4 + 3];
  Qn[n * 3]     = H - al * H;
  Qn[n * 3 + 1] = Ee + be * H - ga * Ee + 0.1f * ec;
  Qn[n * 3 + 2] = V + et * Ee + 0.01f * noise_t[n];
}

// ---------------- denormalize predictions + emit Q_final / P_last ----------------
__global__ void finalize_kernel(const float* __restrict__ xpa,
                                const float* __restrict__ aw, const float* __restrict__ ab,
                                const float* __restrict__ meanb, const float* __restrict__ stdb,
                                const float* __restrict__ Qf, const float* __restrict__ P,
                                float* out) {
  int idx = blockIdx.x * blockDim.x + threadIdx.x;
  if (idx < NN * HOR) {
    int t = idx % HOR; int be = idx / HOR; int e = be & (EE - 1); int b = be >> 7;
    float v = (xpa[(size_t)t * NN + be] - ab[e]) / aw[e];
    out[idx] = v * stdb[b] + meanb[b];
  }
  if (idx < NN * 3) out[NN * HOR + idx] = Qf[idx];
  if (idx < NN * 4) out[NN * HOR + NN * 3 + idx] = P[idx];
}

// ---------------- host driver ----------------
extern "C" void kernel_launch(void* const* d_in, const int* in_sizes, int n_in,
                              void* d_out, int out_size, void* d_ws, size_t ws_size,
                              hipStream_t stream) {
  const float* x    = (const float*)d_in[0];
  const float* nois = (const float*)d_in[1];
  const float* aw   = (const float*)d_in[2];
  const float* ab   = (const float*)d_in[3];
  const float* pw1  = (const float*)d_in[4];
  const float* pb1  = (const float*)d_in[5];
  const float* bng  = (const float*)d_in[6];
  const float* bnb  = (const float*)d_in[7];
  const float* bnm  = (const float*)d_in[8];
  const float* bnv  = (const float*)d_in[9];
  const float* pw2  = (const float*)d_in[10];
  const float* pb2  = (const float*)d_in[11];
  const float* nw1  = (const float*)d_in[12];
  const float* nb1  = (const float*)d_in[13];
  const float* nw2  = (const float*)d_in[14];
  const float* nb2  = (const float*)d_in[15];
  const float* nw3  = (const float*)d_in[16];
  const float* nb3  = (const float*)d_in[17];
  const float* gwl  = (const float*)d_in[18];
  const float* gbl  = (const float*)d_in[19];
  const float* gwr  = (const float*)d_in[20];
  const float* gbr  = (const float*)d_in[21];
  const float* gattn= (const float*)d_in[22];
  const float* outw = (const float*)d_in[23];
  const float* outb = (const float*)d_in[24];
  const float* cew  = (const float*)d_in[25];
  const float* ceb  = (const float*)d_in[26];
  const float* cdw  = (const float*)d_in[27];
  const float* cdb  = (const float*)d_in[28];
  const float* qw   = (const float*)d_in[29];
  const float* qb   = (const float*)d_in[30];
  const float* kw   = (const float*)d_in[31];
  const float* kb   = (const float*)d_in[32];
  float* out = (float*)d_out;

  char* ws = (char*)d_ws;
  size_t off = 0;
  auto alloc = [&](size_t bytes) -> void* {
    void* p = ws + off;
    off += (bytes + 255) & ~(size_t)255;
    return p;
  };
  float*    meanb = (float*)   alloc(BB * 4);
  float*    stdb  = (float*)   alloc(BB * 4);
  float*    xf    = (float*)   alloc((size_t)NN * WW * 4);
  _Float16* xf16  = (_Float16*)alloc((size_t)NN * WW * 2);
  _Float16* qw16  = (_Float16*)alloc(32 * WW * 2);
  _Float16* kw16  = (_Float16*)alloc(32 * WW * 2);
  _Float16* wl16  = (_Float16*)alloc(HID * HID * 2);
  _Float16* wr16  = (_Float16*)alloc(HID * HID * 2);
  _Float16* q16   = (_Float16*)alloc((size_t)NN * 32 * 2);
  _Float16* k16   = (_Float16*)alloc((size_t)NN * 32 * 2);
  float*    logits= (float*)   alloc((size_t)BB * EE * EE * 4);
  float*    adjbin= (float*)   alloc((size_t)BB * EE * EE * 4);
  _Float16* a16   = (_Float16*)alloc((size_t)BB * EE * EE * 2);
  float*    xc    = (float*)   alloc((size_t)NN * WIN * 4);
  float*    hbuf  = (float*)   alloc((size_t)NN * HID * 4);
  _Float16* h16   = (_Float16*)alloc((size_t)NN * HID * 2);
  float*    gl    = (float*)   alloc((size_t)NN * HID * 4);
  float*    gr    = (float*)   alloc((size_t)NN * HID * 4);
  _Float16* gr16  = (_Float16*)alloc((size_t)NN * HID * 2);
  float*    ht    = (float*)   alloc((size_t)NN * HID * 4);
  float*    hn    = (float*)   alloc((size_t)NN * HID * 4);
  float*    QA    = (float*)   alloc((size_t)NN * 3 * 4);
  float*    QB    = (float*)   alloc((size_t)NN * 3 * 4);
  float*    Pbuf  = (float*)   alloc((size_t)NN * 4 * 4);
  float*    xpa   = (float*)   alloc((size_t)HOR * NN * 4);

  float* adj_soft_out = out + NN * HOR + NN * 3 + NN * 4;   // 4th output region

  // ---- prep ----
  stats_kernel  <<<BB, WW, 0, stream>>>(x, meanb, stdb);
  prep_xf_kernel<<<BB, 256, 0, stream>>>(x, meanb, stdb, pw1, pb1, bng, bnb, bnm, bnv,
                                         pw2, pb2, aw, ab, xf, xf16, xc);
  cast16_kernel<<<(32 * WW + 255) / 256, 256, 0, stream>>>(qw, qw16, 32 * WW);
  cast16_kernel<<<(32 * WW + 255) / 256, 256, 0, stream>>>(kw, kw16, 32 * WW);
  cast16_kernel<<<(HID * HID + 255) / 256, 256, 0, stream>>>(gwl, wl16, HID * HID);
  cast16_kernel<<<(HID * HID + 255) / 256, 256, 0, stream>>>(gwr, wr16, HID * HID);

  // ---- adjacency (WMMA GEMMs + softmax) ----
  gemm_qk_kernel    <<<dim3(NN / 16, 2, 2), 32, 0, stream>>>(xf16, qw16, kw16, qb, kb, q16, k16);
  gemm_adj_kernel   <<<dim3(EE / 16, EE / 16, BB), 32, 0, stream>>>(q16, k16, logits);
  adj_softmax_kernel<<<NN, EE, 0, stream>>>(logits, adj_soft_out, adjbin);

  init_q_kernel<<<(NN + 255) / 256, 256, 0, stream>>>(QA);

  // ---- 12-step scan ----
  float* Qcur = QA;
  float* Qnxt = QB;
  for (int t = 0; t < HOR; ++t) {
    cnn_kernel     <<<NN, 256, 0, stream>>>(xc, nw1, nb1, nw2, nb2, nw3, nb3, hbuf, h16, t);
    gemm_gat_kernel<<<dim3(NN / 16, HID / 16, 2), 32, 0, stream>>>(h16, wl16, wr16, gbl, gbr, gl, gr, gr16);
    gat_e_kernel   <<<NN, EE, 0, stream>>>(gl, gr, gattn, adjbin, a16);
    gemm_ht_kernel <<<dim3(EE / 16, HID / 16, BB), 32, 0, stream>>>(a16, gr16, ht);
    combine_kernel <<<(NN * HID) / 256, 256, 0, stream>>>(hbuf, ht, Qcur, cew, ceb, hn);
    pred_p_kernel  <<<NN, HID, 0, stream>>>(hn, outw, outb, cdw, cdb, xpa, xc, Pbuf, t);
    mech_kernel    <<<(NN + 255) / 256, 256, 0, stream>>>(Qcur, Qnxt, Pbuf, adjbin, nois + (size_t)t * NN);
    float* tmp = Qcur; Qcur = Qnxt; Qnxt = tmp;
  }

  // after 12 swaps Qcur points at QA (the final state)
  finalize_kernel<<<(NN * HOR + 255) / 256, 256, 0, stream>>>(xpa, aw, ab, meanb, stdb, Qcur, Pbuf, out);
}
